// VGG16_GAP_64690797413087
// MI455X (gfx1250) — 8-bit matrix operand order repaired and checked by reading, compiled, not run on hardware
//
#include <hip/hip_runtime.h>
#include <stdint.h>

typedef int v16i __attribute__((ext_vector_type(16)));
typedef int v4i __attribute__((ext_vector_type(4)));
typedef float v8f __attribute__((ext_vector_type(8)));

#define BN_EPS 1e-5f
#define BATCH 16
#define FP8_P1 ((uint8_t)0x38)
#define FP8_M1 ((uint8_t)0xB8)

__global__ void binarize_w_kernel(const float* __restrict__ w,
                                  uint8_t* __restrict__ w8,
                                  int cin, int cout, int ks, int K, int Kpad) {
  const int k = blockIdx.x * 256 + threadIdx.x;
  const int o = blockIdx.y;
  if (k >= Kpad) return;
  uint8_t v = 0;
  if (k < K && o < cout) {
    const int lgc = __builtin_ctz(cin);
    const int e = k >> lgc;
    const int c = k & (cin - 1);
    const int dy = e / ks, dx = e % ks;
    float wv = w[(((size_t)o * cin + c) * ks + dy) * ks + dx];
    v = (wv >= 0.0f) ? FP8_P1 : FP8_M1;
  }
  w8[(size_t)o * Kpad + k] = v;
}

__global__ void conv0_direct_kernel(const float* __restrict__ x,
                                    const float* __restrict__ w,
                                    const float* __restrict__ bias,
                                    float* __restrict__ y) {
  const int HW = 128 * 128;
  long long idx = (long long)blockIdx.x * blockDim.x + threadIdx.x;
  if (idx >= (long long)BATCH * 64 * HW) return;
  int p = (int)(idx & (HW - 1));
  int m = (int)((idx >> 14) & 63);
  int img = (int)(idx >> 20);
  int h = p >> 7, wv = p & 127;
  float s = bias[m];
#pragma unroll
  for (int c = 0; c < 3; ++c)
#pragma unroll
    for (int dy = 0; dy < 3; ++dy)
#pragma unroll
      for (int dx = 0; dx < 3; ++dx) {
        int hy = h + dy - 1, wx = wv + dx - 1;
        if (hy < 0 || hy > 127 || wx < 0 || wx > 127) continue;
        float xv = x[(((size_t)img * 3 + c) * 128 + hy) * 128 + wx];
        float ww = w[((m * 3 + c) * 3 + dy) * 3 + dx];
        s += (ww >= 0.0f) ? xv : -xv;
      }
  y[idx] = s;
}

#define LDSR 68
__global__ void bconv3x3_wmma_kernel(const uint8_t* __restrict__ act,
                                     const uint8_t* __restrict__ w8,
                                     const float* __restrict__ bias,
                                     float* __restrict__ y,
                                     int C, int Cout, int lgW) {
  const int W = 1 << lgW;
  const int HW = W << lgW;
  const int lgC = __builtin_ctz(C);
  const int K = 9 * C;
  const int Kpad = (K + 127) & ~127;

  const int lane = threadIdx.x;
  const int wid = threadIdx.y;
  const int tid = wid * 32 + lane;
  const int img = blockIdx.z;
  const int pBase = blockIdx.x * 32;
  const int mBase = (blockIdx.y * 4 + wid) * 16;

  const uint8_t* actI = act + (size_t)img * HW * C;
  float* yI = y + (size_t)img * Cout * HW;

  __shared__ int ldsB[32 * LDSR];

  const int half = lane >> 4;
  const int l16 = lane & 15;

  int kA[16], kB[16];
#pragma unroll
  for (int v = 0; v < 16; ++v) {
    kA[v] = (v / 4) * 32 + ((v % 4) / 2) * 16 + half * 8 + (v % 2) * 4;
    kB[v] = (v / 4) * 32 + ((v % 4) / 2) * 16 + half * 8 + (v % 2) * 4;
  }

  v8f acc0 = {}, acc1 = {};
  const int* w32 = (const int*)w8;
  const int nKB = Kpad >> 7;

  for (int kb = 0; kb < nKB; ++kb) {
    const int koff = kb << 7;
    __syncthreads();
#pragma unroll
    for (int s = tid; s < 256; s += 128) {
      const int n = s >> 3, q = s & 7;
      const int p = pBase + n;
      const int ph = p >> lgW, pw = p & (W - 1);
      const int kq = koff + q * 16;
      v4i val = {};
      const int e = kq >> lgC;
      if (e < 9) {
        const int dy = e / 3 - 1, dx = e % 3 - 1;
        const int hy = ph + dy, wx = pw + dx;
        if (((unsigned)hy < (unsigned)W) & ((unsigned)wx < (unsigned)W)) {
          const int c = kq & (C - 1);
          val = *(const v4i*)(actI + ((size_t)((hy << lgW) + wx)) * C + c);
        }
      }
      *(v4i*)&ldsB[n * LDSR + q * 4] = val;
    }
    __syncthreads();

    const int rowBase = ((mBase + l16) * Kpad + koff) >> 2;
    __builtin_prefetch(&w32[rowBase + 32], 0, 1);
    v16i a, b0, b1;
#pragma unroll
    for (int v = 0; v < 16; ++v) a[v] = w32[rowBase + (kA[v] >> 2)];
#pragma unroll
    for (int v = 0; v < 16; ++v) b0[v] = ldsB[l16 * LDSR + (kB[v] >> 2)];
#pragma unroll
    for (int v = 0; v < 16; ++v) b1[v] = ldsB[(l16 + 16) * LDSR + (kB[v] >> 2)];

    acc0 = __builtin_amdgcn_wmma_f32_16x16x128_fp8_fp8(a, b0, (short)0, acc0, false, false);
    acc1 = __builtin_amdgcn_wmma_f32_16x16x128_fp8_fp8(a, b1, (short)0, acc1, false, false);
  }

#pragma unroll
  for (int r = 0; r < 8; ++r) {
    const int m = mBase + half * 8 + r;
    const float bv = bias[m];
    yI[(size_t)m * HW + pBase + l16] = acc0[r] + bv;
    yI[(size_t)m * HW + pBase + 16 + l16] = acc1[r] + bv;
  }
}

__global__ void bgemm_wmma_kernel(const uint8_t* __restrict__ act,
                                  const uint8_t* __restrict__ w8,
                                  float* __restrict__ y,
                                  int K, int Npix) {
  const int lane = threadIdx.x;
  const int wid = threadIdx.y;
  const int tid = wid * 32 + lane;
  const int tileN = blockIdx.x;
  const int mBase = (blockIdx.y * blockDim.y + wid) * 16;

  __shared__ int ldsB[16 * LDSR];
  const int half = lane >> 4;
  const int l16 = lane & 15;

  int kA[16], kB[16];
#pragma unroll
  for (int v = 0; v < 16; ++v) {
    kA[v] = (v / 4) * 32 + ((v % 4) / 2) * 16 + half * 8 + (v % 2) * 4;
    kB[v] = (v / 4) * 32 + ((v % 4) / 2) * 16 + half * 8 + (v % 2) * 4;
  }

  const int n = tid >> 3, q = tid & 7;
  const unsigned long long actBase = (unsigned long long)(uintptr_t)act;

  v8f acc = {};
  const int nKB = K >> 7;
  const int* w32 = (const int*)w8;

  for (int kb = 0; kb < nKB; ++kb) {
    const int koff = kb << 7;
    __syncthreads();
    {
      unsigned lds_off = (unsigned)(uintptr_t)&ldsB[n * LDSR + q * 4];
      unsigned goff = (unsigned)((tileN * 16 + n) * K + koff + q * 16);
      asm volatile("global_load_async_to_lds_b128 %0, %1, %2"
                   :
                   : "v"(lds_off), "v"(goff), "s"(actBase)
                   : "memory");
    }
    asm volatile("s_wait_asynccnt 0x0" ::: "memory");
    __syncthreads();

    const int rowBase = ((mBase + l16) * K + koff) >> 2;
    v16i a, b;
#pragma unroll
    for (int v = 0; v < 16; ++v) a[v] = w32[rowBase + (kA[v] >> 2)];
#pragma unroll
    for (int v = 0; v < 16; ++v) b[v] = ldsB[l16 * LDSR + (kB[v] >> 2)];

    acc = __builtin_amdgcn_wmma_f32_16x16x128_fp8_fp8(a, b, (short)0, acc, false, false);
  }

#pragma unroll
  for (int r = 0; r < 8; ++r) {
    const int m = mBase + half * 8 + r;
    y[(size_t)m * Npix + tileN * 16 + l16] = acc[r];
  }
}

__global__ void bn_pool_sign_kernel(const float* __restrict__ y,
                                    const float* __restrict__ g,
                                    const float* __restrict__ bb,
                                    const float* __restrict__ mm,
                                    const float* __restrict__ vv,
                                    uint8_t* __restrict__ out,
                                    int Cout, int H, int W, int pool) {
  const int Ho = pool ? H >> 1 : H;
  const int Wo = pool ? W >> 1 : W;
  const int lgC = __builtin_ctz(Cout);
  const int lgWo = __builtin_ctz(Wo);
  const int lgHo = __builtin_ctz(Ho);
  const long long total = (long long)BATCH * Ho * Wo * Cout;
  long long idx = (long long)blockIdx.x * blockDim.x + threadIdx.x;
  if (idx >= total) return;
  const int c = (int)(idx & (Cout - 1));
  long long r = idx >> lgC;
  const int wo = (int)(r & (Wo - 1));
  r >>= lgWo;
  const int ho = (int)(r & (Ho - 1));
  const int img = (int)(r >> lgHo);
  const float* yc = y + ((size_t)img * Cout + c) * H * W;
  float val;
  if (pool) {
    int h0 = ho * 2, w0 = wo * 2;
    float a = yc[h0 * W + w0], b2 = yc[h0 * W + w0 + 1];
    float c2 = yc[(h0 + 1) * W + w0], d = yc[(h0 + 1) * W + w0 + 1];
    val = fmaxf(fmaxf(a, b2), fmaxf(c2, d));
  } else {
    val = yc[ho * W + wo];
  }
  float scale = g[c] * rsqrtf(vv[c] + BN_EPS);
  float h2 = (val - mm[c]) * scale + bb[c];
  h2 = fminf(1.0f, fmaxf(-1.0f, h2));
  out[idx] = (h2 >= 0.0f) ? FP8_P1 : FP8_M1;
}

__global__ void gap_cls_kernel(const float* __restrict__ y13,
                               const float* __restrict__ g,
                               const float* __restrict__ bb,
                               const float* __restrict__ mm,
                               const float* __restrict__ vv,
                               const float* __restrict__ cm,
                               const float* __restrict__ cv,
                               float* __restrict__ zbuf) {
  int idx = blockIdx.x * blockDim.x + threadIdx.x;
  if (idx >= BATCH * 1000) return;
  int c = idx % 1000;
  int img = idx / 1000;
  float scale = g[c] * rsqrtf(vv[c] + BN_EPS);
  float sh = bb[c] - mm[c] * scale;
  float s = 0.0f;
#pragma unroll
  for (int p = 0; p < 16; ++p) {
    float hv = y13[(size_t)c * 256 + img * 16 + p] * scale + sh;
    hv = fminf(1.0f, fmaxf(-1.0f, hv));
    s += (hv >= 0.0f) ? 1.0f : -1.0f;
  }
  s *= (1.0f / 16.0f);
  zbuf[idx] = (s - cm[c]) * rsqrtf(cv[c] + BN_EPS);
}

__global__ void logsoftmax_kernel(const float* __restrict__ z,
                                  float* __restrict__ out) {
  __shared__ float red[256];
  const int img = blockIdx.x;
  const int tid = threadIdx.x;
  const float* zi = z + (size_t)img * 1000;

  float mx = -1e30f;
  for (int c = tid; c < 1000; c += 256) mx = fmaxf(mx, zi[c]);
  red[tid] = mx;
  __syncthreads();
  for (int s = 128; s > 0; s >>= 1) {
    if (tid < s) red[tid] = fmaxf(red[tid], red[tid + s]);
    __syncthreads();
  }
  mx = red[0];
  __syncthreads();

  float sum = 0.0f;
  for (int c = tid; c < 1000; c += 256) sum += expf(zi[c] - mx);
  red[tid] = sum;
  __syncthreads();
  for (int s = 128; s > 0; s >>= 1) {
    if (tid < s) red[tid] += red[tid + s];
    __syncthreads();
  }
  float lse = logf(red[0]) + mx;

  for (int c = tid; c < 1000; c += 256) out[(size_t)img * 1000 + c] = zi[c] - lse;
}

static const int CIN_[14] = {3, 64, 64, 128, 128, 256, 256, 256, 512, 512, 512, 512, 512, 512};
static const int COUT_[14] = {64, 64, 128, 128, 256, 256, 256, 512, 512, 512, 512, 512, 512, 1000};
static const int POOL_[14] = {0, 1, 0, 1, 0, 0, 1, 0, 0, 1, 0, 0, 1, 0};

extern "C" void kernel_launch(void* const* d_in, const int* in_sizes, int n_in,
                              void* d_out, int out_size, void* d_ws, size_t ws_size,
                              hipStream_t stream) {
  (void)in_sizes; (void)n_in; (void)out_size; (void)ws_size;

  const float* x = (const float*)d_in[0];
  const float* conv_w[14];
  for (int i = 0; i < 14; ++i) conv_w[i] = (const float*)d_in[1 + i];
  const float* conv_b[13];
  for (int i = 0; i < 13; ++i) conv_b[i] = (const float*)d_in[15 + i];
  const float* bng[14]; const float* bnb[14]; const float* bnm[14]; const float* bnv[14];
  for (int i = 0; i < 14; ++i) {
    bng[i] = (const float*)d_in[28 + i];
    bnb[i] = (const float*)d_in[42 + i];
    bnm[i] = (const float*)d_in[56 + i];
    bnv[i] = (const float*)d_in[70 + i];
  }
  const float* cls_m = (const float*)d_in[84];
  const float* cls_v = (const float*)d_in[85];
  float* out = (float*)d_out;

  uint8_t* ws = (uint8_t*)d_ws;
  size_t cur = 0;
  auto carve = [&](size_t bytes) -> uint8_t* {
    uint8_t* p = ws + cur;
    cur = (cur + bytes + 255) & ~(size_t)255;
    return p;
  };

  uint8_t* w8[14];
  int Mpad[14], Kw[14], Kp[14];
  for (int i = 1; i <= 13; ++i) {
    int ks = (i < 13) ? 3 : 1;
    Kw[i] = ks * ks * CIN_[i];
    Kp[i] = (Kw[i] + 127) & ~127;
    Mpad[i] = (i < 13) ? COUT_[i] : 1024;
    w8[i] = carve((size_t)Mpad[i] * Kp[i]);
  }
  uint8_t* actA = carve((size_t)BATCH * 128 * 128 * 64);
  uint8_t* actB = carve((size_t)BATCH * 128 * 128 * 64);
  float* yBuf = (float*)carve((size_t)BATCH * 64 * 128 * 128 * sizeof(float));
  float* y13 = (float*)carve((size_t)1024 * 256 * sizeof(float));
  float* zbuf = (float*)carve((size_t)BATCH * 1000 * sizeof(float));

  for (int i = 1; i <= 13; ++i) {
    int ks = (i < 13) ? 3 : 1;
    dim3 grid((Kp[i] + 255) / 256, Mpad[i]);
    binarize_w_kernel<<<grid, 256, 0, stream>>>(conv_w[i], w8[i], CIN_[i], COUT_[i], ks, Kw[i], Kp[i]);
  }

  {
    long long total = (long long)BATCH * 64 * 128 * 128;
    conv0_direct_kernel<<<(int)((total + 255) / 256), 256, 0, stream>>>(x, conv_w[0], conv_b[0], yBuf);
    bn_pool_sign_kernel<<<(int)((total + 255) / 256), 256, 0, stream>>>(
        yBuf, bng[0], bnb[0], bnm[0], bnv[0], actA, 64, 128, 128, 0);
  }

  uint8_t* actIn = actA;
  uint8_t* actOut = actB;
  int Hc = 128;
  for (int i = 1; i <= 12; ++i) {
    const int C = CIN_[i], Co = COUT_[i], H = Hc, W = Hc, HW = H * W;
    int lgW = __builtin_ctz(W);
    dim3 block(32, 4);
    dim3 grid(HW / 32, Co / 64, BATCH);
    bconv3x3_wmma_kernel<<<grid, block, 0, stream>>>(actIn, w8[i], conv_b[i], yBuf, C, Co, lgW);

    const int pool = POOL_[i];
    const int Ho = pool ? H / 2 : H;
    long long total = (long long)BATCH * Ho * Ho * Co;
    bn_pool_sign_kernel<<<(int)((total + 255) / 256), 256, 0, stream>>>(
        yBuf, bng[i], bnb[i], bnm[i], bnv[i], actOut, Co, H, W, pool);

    uint8_t* t = actIn; actIn = actOut; actOut = t;
    if (pool) Hc /= 2;
  }

  {
    dim3 block(32, 4);
    dim3 grid(256 / 16, 1024 / 64, 1);
    bgemm_wmma_kernel<<<grid, block, 0, stream>>>(actIn, w8[13], y13, 512, 256);
  }

  gap_cls_kernel<<<(BATCH * 1000 + 255) / 256, 256, 0, stream>>>(
      y13, bng[13], bnb[13], bnm[13], bnv[13], cls_m, cls_v, zbuf);
  logsoftmax_kernel<<<BATCH, 256, 0, stream>>>(zbuf, out);
}
